// GaussianTrans_14224931684887
// MI455X (gfx1250) — matrix operand order repaired and checked by reading, not compiled, not run on hardware
//
#include <hip/hip_runtime.h>
#include <hip/hip_bf16.h>

// GaussianTrans fused dual-axis attention for MI455X (gfx1250, wave32).
// Pass ROW: out = softmax_w(ax[b,r,c,w] + D[c,w]) @ value[b,r,w,:]
// Pass COL: out += softmax_h(ay[b,c,r,h] + D[r,h]) @ value[b,h,c,:]
// GEMMs via v_wmma_f32_16x16x32_bf16; probabilities + V staged in LDS as bf16.

#define S  192
#define C  64
#define ST 200            // LDS row stride in bf16 elements (100 dwords; gcd(36,64)=4 -> 16 distinct banks)
#define NWAVE 6
#define BLOCK (NWAVE * 32)

typedef __attribute__((ext_vector_type(16))) __bf16 v16bf;
typedef __attribute__((ext_vector_type(8)))  float  v8f;

__device__ __forceinline__ unsigned short f2bf(float f) {
  unsigned int u = __builtin_bit_cast(unsigned int, f);
  u += 0x7FFFu + ((u >> 16) & 1u);        // round-to-nearest-even
  return (unsigned short)(u >> 16);
}

__device__ __forceinline__ unsigned int lds32(const unsigned short* p) {
  return *reinterpret_cast<const unsigned int*>(p);   // -> ds_load_b32
}

template <bool COLPASS>
__global__ __launch_bounds__(BLOCK)
void gt_pass(const float* __restrict__ attn, const float* __restrict__ value,
             const float* __restrict__ shiftp, const float* __restrict__ biasp,
             float* __restrict__ out)
{
  __shared__ unsigned short Vt[C * ST];            // value^T tile, bf16: Vt[d][k]
  __shared__ unsigned short Pst[NWAVE * 16 * ST];  // per-wave probability staging, bf16

  const int tid  = threadIdx.x;
  const int lane = tid & 31;
  const int wave = tid >> 5;

  const int p = blockIdx.x;          // (b, r) for ROW pass, (b, c) for COL pass
  const int b = p / S;
  const int i = p - b * S;

  const float shift = shiftp[0];
  const float nbias = -biasp[0];

  const float* attnb = attn + (size_t)p * (S * S);
  const size_t base  = COLPASS ? ((size_t)b * ((size_t)S * S * C) + (size_t)i * C)
                               : ((size_t)p * (S * C));
  const float* vb = value + base;
  float*       ob = out   + base;
  const int stride = COLPASS ? (S * C) : C;        // k-row stride of V and m-row stride of out

  // ---- stage value^T into LDS as bf16 (coalesced float4 reads) ----
  for (int e = tid * 4; e < S * C; e += BLOCK * 4) {
    const int k = e >> 6;            // C == 64
    const int d = e & 63;
    const float4 f = *reinterpret_cast<const float4*>(vb + (size_t)k * stride + d);
    Vt[(d + 0) * ST + k] = f2bf(f.x);
    Vt[(d + 1) * ST + k] = f2bf(f.y);
    Vt[(d + 2) * ST + k] = f2bf(f.z);
    Vt[(d + 3) * ST + k] = f2bf(f.w);
  }
  __syncthreads();

  unsigned short* Pw = Pst + wave * (16 * ST);
  const v8f zero8 = {0.f, 0.f, 0.f, 0.f, 0.f, 0.f, 0.f, 0.f};
  const int mrow = lane & 15;
  const int hi   = lane >> 4;

  for (int t = wave; t < S / 16; t += NWAVE) {     // 12 tiles, 2 per wave, no divergence
    const int m0 = t * 16;

    // ---- softmax of 16 rows (whole wave per row; coalesced 128B reads) ----
    for (int rr = 0; rr < 16; ++rr) {
      const int m = m0 + rr;
      const float* ar = attnb + (size_t)m * S;
      float v[6];
      float mx = -3.4e38f;
#pragma unroll
      for (int j = 0; j < 6; ++j) {
        const int k = lane + 32 * j;
        const float dk = (float)(k - m);
        const float logit = ar[k] + __builtin_fmaf(-shift, dk * dk, nbias); // + D[m][k]
        v[j] = logit;
        mx = fmaxf(mx, logit);
      }
#pragma unroll
      for (int off = 16; off >= 1; off >>= 1) mx = fmaxf(mx, __shfl_xor(mx, off, 32));
      float s = 0.f;
#pragma unroll
      for (int j = 0; j < 6; ++j) { v[j] = __expf(v[j] - mx); s += v[j]; }
#pragma unroll
      for (int off = 16; off >= 1; off >>= 1) s += __shfl_xor(s, off, 32);
      const float inv = 1.0f / s;
#pragma unroll
      for (int j = 0; j < 6; ++j)
        Pw[rr * ST + lane + 32 * j] = f2bf(v[j] * inv);
    }
    __builtin_amdgcn_wave_barrier();   // same-wave LDS ops are in-order; fence the compiler

    // ---- [16 x 192] @ [192 x 64] via bf16 WMMA, f32 accumulate ----
    v8f acc[4] = {zero8, zero8, zero8, zero8};
    for (int kt = 0; kt < 6; ++kt) {
      const int kb = kt * 32;
      // A fragment (16x32 bf16): VGPR j<4: K = 2j,2j+1 (+8 for lanes 16-31); j>=4: +16
      union { v16bf v; unsigned int u[8]; } A;
      const int ka = kb + hi * 8;
#pragma unroll
      for (int j = 0; j < 8; ++j) {
        const int K = ka + ((j >> 2) << 4) + ((j & 3) << 1);
        A.u[j] = lds32(&Pw[mrow * ST + K]);
      }
      const int kbB = kb + (hi << 3);  // B (32x16): same K order as A (K 0..7,16..23 lanes 0-15; 8..15,24..31 lanes 16-31)
#pragma unroll
      for (int n = 0; n < 4; ++n) {
        union { v16bf v; unsigned int u[8]; } Bf;
        const int dcol = (n << 4) + mrow;
#pragma unroll
        for (int j = 0; j < 8; ++j)
          Bf.u[j] = lds32(&Vt[dcol * ST + kbB + ((j >> 2) << 4) + ((j & 3) << 1)]);
        acc[n] = __builtin_amdgcn_wmma_f32_16x16x32_bf16(
            false, A.v, false, Bf.v, (short)0, acc[n], false, false);
      }
    }

    // ---- write / accumulate 16x64 output tile ----
    const int rbase = m0 + hi * 8;     // acc VGPR q: row m0 + q (lanes 0-15), m0+8+q (16-31)
#pragma unroll
    for (int n = 0; n < 4; ++n) {
      const int d = (n << 4) + mrow;
#pragma unroll
      for (int q = 0; q < 8; ++q) {
        float* op = ob + (size_t)(rbase + q) * stride + d;
        if (COLPASS) *op += acc[n][q]; else *op = acc[n][q];
      }
    }
    __builtin_amdgcn_wave_barrier();   // protect Pw WAR before next tile
  }
}

extern "C" void kernel_launch(void* const* d_in, const int* in_sizes, int n_in,
                              void* d_out, int out_size, void* d_ws, size_t ws_size,
                              hipStream_t stream) {
  (void)in_sizes; (void)n_in; (void)out_size; (void)d_ws; (void)ws_size;
  // inputs: 0:x (unused), 1:atten_x_full, 2:atten_y_full, 3:value_full, 4:shift, 5:bias
  const float* ax  = (const float*)d_in[1];
  const float* ay  = (const float*)d_in[2];
  const float* val = (const float*)d_in[3];
  const float* shf = (const float*)d_in[4];
  const float* bia = (const float*)d_in[5];
  float* out = (float*)d_out;

  const int B = 4;
  dim3 grid(B * S), block(BLOCK);
  gt_pass<false><<<grid, block, 0, stream>>>(ax, val, shf, bia, out);  // out = row attention
  gt_pass<true ><<<grid, block, 0, stream>>>(ay, val, shf, bia, out);  // out += col attention
}
